// FiLMSelfAttention_77275051590259
// MI455X (gfx1250) — compile-verified, matrix operand order checked by reading, not run on hardware
//
#include <hip/hip_runtime.h>
#include <hip/hip_bf16.h>

// ---------------------------------------------------------------------------
// FiLM self-attention for MI455X (gfx1250, wave32, WMMA).
// Shapes: B=4, N=128, ND=256, ED=128, FD=128, H=8, DF=32.
// Big GEMMs run on v_wmma_f32_16x16x32_f16 (f16 operands, fp32 accumulate).
// ---------------------------------------------------------------------------

#define BB 4
#define NN 128
#define NDD 256
#define EDD 128
#define FDD 128

typedef __attribute__((ext_vector_type(16))) _Float16 v16h;
typedef __attribute__((ext_vector_type(8)))  float    v8f;

__device__ __forceinline__ v8f zero8() {
  v8f z;
#pragma unroll
  for (int q = 0; q < 8; ++q) z[q] = 0.0f;
  return z;
}

// k-index inside a 16x32 f16 A-fragment (and symmetric B-fragment):
// lane 0-15 hold K 0..7 / 16..23, lanes 16-31 hold K 8..15 / 24..31.
__device__ __forceinline__ int wmma_kidx(int lane, int h) {
  return ((h >> 3) << 4) + ((lane >> 4) << 3) + (h & 7);
}

// ---------------------------------------------------------------------------
// Pack the three big weight matrices (f32, row-major (K, Nout)) into per-lane
// WMMA B fragments (f16). Frag f: 512 halves, lane L's 16 halves contiguous.
//   frags 0..63    : edges_nodes_mul  (K=128, N=256)  kt*16+nt
//   frags 64..127  : edges_nodes_add  (K=128, N=256)
//   frags 128..191 : edges_net        (K=256, N=128)  kt*8+nt
// ---------------------------------------------------------------------------
__global__ __launch_bounds__(32) void prepack_kernel(
    const float* __restrict__ wmul, const float* __restrict__ wadd,
    const float* __restrict__ wnet, _Float16* __restrict__ pack) {
  int f = blockIdx.x;
  int lane = threadIdx.x;
  const float* W;
  int Nout, fl;
  _Float16* dst;
  if (f < 64)       { W = wmul; Nout = 256; fl = f;       dst = pack; }
  else if (f < 128) { W = wadd; Nout = 256; fl = f - 64;  dst = pack + 32768; }
  else              { W = wnet; Nout = 128; fl = f - 128; dst = pack + 65536; }
  int ntc = (Nout == 256) ? 16 : 8;
  int kt = fl / ntc, nt = fl % ntc;
#pragma unroll
  for (int h = 0; h < 16; ++h) {
    int k = kt * 32 + wmma_kidx(lane, h);
    int n = nt * 16 + (lane & 15);
    dst[(size_t)fl * 512 + lane * 16 + h] = (_Float16)W[(size_t)k * Nout + n];
  }
}

// ---------------------------------------------------------------------------
// Q/K/V projections (fp32, masked). One block per (b,i) row, 256 threads.
// ---------------------------------------------------------------------------
__global__ __launch_bounds__(256) void qkv_kernel(
    const float* __restrict__ nodes, const float* __restrict__ wq,
    const float* __restrict__ bq, const float* __restrict__ wk,
    const float* __restrict__ bk, const float* __restrict__ wv,
    const float* __restrict__ bv, const float* __restrict__ mask,
    float* __restrict__ Q, float* __restrict__ K, float* __restrict__ V) {
  int row = blockIdx.x;  // b*N + i
  int o = threadIdx.x;
  __shared__ float xs[NDD];
  xs[o] = nodes[(size_t)row * NDD + o];
  __syncthreads();
  float m = mask[row];
  float aq = bq[o], ak = bk[o], av = bv[o];
  for (int i = 0; i < NDD; ++i) {
    float x = xs[i];
    aq += x * wq[(size_t)i * NDD + o];
    ak += x * wk[(size_t)i * NDD + o];
    av += x * wv[(size_t)i * NDD + o];
  }
  Q[(size_t)row * NDD + o] = aq * m;
  K[(size_t)row * NDD + o] = ak * m;
  V[(size_t)row * NDD + o] = av * m;
}

// ---------------------------------------------------------------------------
// Per-batch FiLM vectors from features: fe_mul, fe_add, fn_mul, fn_add.
// ---------------------------------------------------------------------------
__global__ __launch_bounds__(256) void fvec_kernel(
    const float* __restrict__ features,
    const float* __restrict__ wem, const float* __restrict__ bem,
    const float* __restrict__ wea, const float* __restrict__ bea,
    const float* __restrict__ wnm, const float* __restrict__ bnm,
    const float* __restrict__ wna, const float* __restrict__ bna,
    float* __restrict__ feM, float* __restrict__ feA,
    float* __restrict__ fnM, float* __restrict__ fnA) {
  int b = blockIdx.x;
  int t = threadIdx.x;
  __shared__ float fsh[FDD];
  if (t < FDD) fsh[t] = features[b * FDD + t];
  __syncthreads();
  float s1 = bem[t], s2 = bea[t], s3 = bnm[t], s4 = bna[t];
  for (int k = 0; k < FDD; ++k) {
    float f = fsh[k];
    s1 += f * wem[(size_t)k * NDD + t];
    s2 += f * wea[(size_t)k * NDD + t];
    s3 += f * wnm[(size_t)k * NDD + t];
    s4 += f * wna[(size_t)k * NDD + t];
  }
  feM[b * NDD + t] = s1;
  feA[b * NDD + t] = s2;
  fnM[b * NDD + t] = s3;
  fnA[b * NDD + t] = s4;
}

// ---------------------------------------------------------------------------
// Edge statistics partials per (b,i): sum / min / max / sumsq per channel.
// ---------------------------------------------------------------------------
__global__ __launch_bounds__(128) void stats_edges_kernel(
    const float* __restrict__ edges, float* __restrict__ part) {
  int blk = blockIdx.x;  // b*N + i
  int e = threadIdx.x;
  const float* p = edges + (size_t)blk * NN * EDD + e;
  float s = 0.f, ss = 0.f, mn = __builtin_inff(), mx = -__builtin_inff();
  for (int j = 0; j < NN; ++j) {
    float x = p[(size_t)j * EDD];
    s += x; ss += x * x;
    mn = fminf(mn, x); mx = fmaxf(mx, x);
  }
  float* q = part + (size_t)blk * 512;
  q[e] = s; q[128 + e] = mn; q[256 + e] = mx; q[384 + e] = ss;
}

// ---------------------------------------------------------------------------
// Main WMMA kernel: per (b,i), full j-row of 128 edges.
//   E1 = edges@Wmul+b, E2 = edges@Wadd+b (masked)  -> QK FiLM -> logits (f16)
//   new_edges_pre -> LDS -> edges_net WMMA (K-split accumulate) -> new_edges
// 8 waves, wave mt owns rows j in [mt*16, mt*16+16) -> LDS tile is wave-private
// (no __syncthreads needed).
// ---------------------------------------------------------------------------
__global__ __launch_bounds__(256) void edges_main_kernel(
    const float* __restrict__ edges, const float* __restrict__ Qm,
    const float* __restrict__ Km, const float* __restrict__ mask,
    const float* __restrict__ feA, const float* __restrict__ feM,
    const float* __restrict__ bias_mul, const float* __restrict__ bias_add,
    const float* __restrict__ bias_net,
    const _Float16* __restrict__ packMul, const _Float16* __restrict__ packAdd,
    const _Float16* __restrict__ packNet,
    _Float16* __restrict__ logits, float* __restrict__ out_edges) {
  int blk = blockIdx.x;        // b*N + i
  int b = blk >> 7;
  int lane = threadIdx.x & 31;
  int mt = threadIdx.x >> 5;   // wave id == M-tile id
  int koff = (lane >> 4) << 3; // 0 or 8

  __shared__ _Float16 As2[128 * 64];  // 16 KB, wave-private stripes

  float maskI = mask[blk];

  // ---- A fragments for the whole K range of this wave's 16 rows -----------
  int jrow = mt * 16 + (lane & 15);
  const float* erow = edges + ((size_t)blk * NN + jrow) * EDD;
  __builtin_prefetch(erow, 0, 1);
  v16h aedge[4];
#pragma unroll
  for (int kt = 0; kt < 4; ++kt) {
    const float* p = erow + kt * 32 + koff;
    float4 x0 = *(const float4*)(p);
    float4 x1 = *(const float4*)(p + 4);
    float4 x2 = *(const float4*)(p + 16);
    float4 x3 = *(const float4*)(p + 20);
    v16h a;
    a[0] = (_Float16)x0.x;  a[1] = (_Float16)x0.y;
    a[2] = (_Float16)x0.z;  a[3] = (_Float16)x0.w;
    a[4] = (_Float16)x1.x;  a[5] = (_Float16)x1.y;
    a[6] = (_Float16)x1.z;  a[7] = (_Float16)x1.w;
    a[8] = (_Float16)x2.x;  a[9] = (_Float16)x2.y;
    a[10] = (_Float16)x2.z; a[11] = (_Float16)x2.w;
    a[12] = (_Float16)x3.x; a[13] = (_Float16)x3.y;
    a[14] = (_Float16)x3.z; a[15] = (_Float16)x3.w;
    aedge[kt] = a;
  }

  v8f neacc[8];
#pragma unroll
  for (int q = 0; q < 8; ++q) neacc[q] = zero8();

  const float inv_sqrt_df = 0.17677669529663687f;  // 1/sqrt(32)

  for (int chunk = 0; chunk < 4; ++chunk) {
    v8f e1acc[4], e2acc[4];
#pragma unroll
    for (int q = 0; q < 4; ++q) { e1acc[q] = zero8(); e2acc[q] = zero8(); }

    // ---- E1/E2 GEMM over K=128 --------------------------------------------
#pragma unroll
    for (int kt = 0; kt < 4; ++kt) {
#pragma unroll
      for (int nt = 0; nt < 4; ++nt) {
        int ntg = chunk * 4 + nt;
        v16h bm = *(const v16h*)(packMul + ((size_t)(kt * 16 + ntg)) * 512 + lane * 16);
        e1acc[nt] = __builtin_amdgcn_wmma_f32_16x16x32_f16(
            false, aedge[kt], false, bm, (short)0, e1acc[nt], false, false);
        v16h ba = *(const v16h*)(packAdd + ((size_t)(kt * 16 + ntg)) * 512 + lane * 16);
        e2acc[nt] = __builtin_amdgcn_wmma_f32_16x16x32_f16(
            false, aedge[kt], false, ba, (short)0, e2acc[nt], false, false);
      }
    }

    // ---- fused epilogue: bias, mask, QK FiLM, logits, features FiLM -------
    int cbase = chunk * 64;
#pragma unroll
    for (int nt = 0; nt < 4; ++nt) {
      int c = cbase + nt * 16 + (lane & 15);
      float qv = Qm[(size_t)blk * NDD + c];
      float bm_ = bias_mul[c], ba_ = bias_add[c];
      float fA = feA[b * NDD + c], fM = feM[b * NDD + c];
#pragma unroll
      for (int r = 0; r < 8; ++r) {
        int j = mt * 16 + r + ((lane >> 4) << 3);
        float mij = maskI * mask[b * NN + j];
        float e1 = (e1acc[nt][r] + bm_) * mij;
        float e2 = (e2acc[nt][r] + ba_) * mij;
        float kv = Km[((size_t)b * NN + j) * NDD + c];
        float lg = (qv * kv * inv_sqrt_df) * (e1 + 1.0f) + e2;
        logits[((size_t)blk * NN + j) * NDD + c] = (_Float16)lg;
        As2[j * 64 + nt * 16 + (lane & 15)] = (_Float16)(fA + (fM + 1.0f) * lg);
      }
    }

    // ---- edges_net GEMM: K-split accumulate over this 64-col chunk --------
#pragma unroll
    for (int kt2 = 0; kt2 < 2; ++kt2) {
      const _Float16* lp = &As2[(mt * 16 + (lane & 15)) * 64 + kt2 * 32 + koff];
      v16h a2;
#pragma unroll
      for (int h = 0; h < 8; ++h) { a2[h] = lp[h]; a2[h + 8] = lp[16 + h]; }
      int ktg = chunk * 2 + kt2;
#pragma unroll
      for (int nt2 = 0; nt2 < 8; ++nt2) {
        v16h bw = *(const v16h*)(packNet + ((size_t)(ktg * 8 + nt2)) * 512 + lane * 16);
        neacc[nt2] = __builtin_amdgcn_wmma_f32_16x16x32_f16(
            false, a2, false, bw, (short)0, neacc[nt2], false, false);
      }
    }
  }

  // ---- write new_edges ----------------------------------------------------
#pragma unroll
  for (int nt2 = 0; nt2 < 8; ++nt2) {
    int c = nt2 * 16 + (lane & 15);
    float bn = bias_net[c];
#pragma unroll
    for (int r = 0; r < 8; ++r) {
      int j = mt * 16 + r + ((lane >> 4) << 3);
      float val = (neacc[nt2][r] + bn) * maskI * mask[b * NN + j];
      out_edges[((size_t)blk * NN + j) * EDD + c] = val;
    }
  }
}

// ---------------------------------------------------------------------------
// Online softmax over j per channel + weighted V + nodes FiLM + nodes_net.
// One block per (b,i), thread c owns channel c.
// ---------------------------------------------------------------------------
__global__ __launch_bounds__(256) void softmax_v_kernel(
    const _Float16* __restrict__ logits, const float* __restrict__ V,
    const float* __restrict__ mask, const float* __restrict__ fnA,
    const float* __restrict__ fnM, const float* __restrict__ wnn,
    const float* __restrict__ bnn, float* __restrict__ out_nodes) {
  int blk = blockIdx.x;  // b*N + i
  int b = blk >> 7;
  int c = threadIdx.x;
  const _Float16* lrow = logits + (size_t)blk * NN * NDD + c;
  float m = -__builtin_inff(), s = 0.f, acc = 0.f;
  for (int j = 0; j < NN; ++j) {
    float l = (mask[b * NN + j] == 0.0f) ? -1e30f : (float)lrow[(size_t)j * NDD];
    float mn = fmaxf(m, l);
    float corr = __expf(m - mn);
    float w = __expf(l - mn);
    s = s * corr + w;
    acc = acc * corr + w * V[((size_t)b * NN + j) * NDD + c];
    m = mn;
  }
  float wv = acc / s;
  float pre = fnA[b * NDD + c] + (fnM[b * NDD + c] + 1.0f) * wv;
  __shared__ float ps[NDD];
  ps[c] = pre;
  __syncthreads();
  float a = bnn[c];
  for (int k = 0; k < NDD; ++k) a += ps[k] * wnn[(size_t)k * NDD + c];
  out_nodes[(size_t)blk * NDD + c] = a * mask[blk];
}

// ---------------------------------------------------------------------------
// Features path: edge-stat reduce, node stats, e2f/n2f/f2f, 2-layer MLP.
// One block per batch b.
// ---------------------------------------------------------------------------
__global__ __launch_bounds__(256) void feat_final_kernel(
    const float* __restrict__ part, const float* __restrict__ nodes,
    const float* __restrict__ features,
    const float* __restrict__ wf2f, const float* __restrict__ bf2f,
    const float* __restrict__ wn2f, const float* __restrict__ bn2f,
    const float* __restrict__ we2f, const float* __restrict__ be2f,
    const float* __restrict__ w1, const float* __restrict__ b1,
    const float* __restrict__ w2, const float* __restrict__ b2,
    float* __restrict__ out_feat) {
  int b = blockIdx.x;
  int t = threadIdx.x;
  __shared__ float ez[512];
  __shared__ float nz[1024];
  __shared__ float fsh[FDD];
  __shared__ float pre[FDD];
  __shared__ float h1[FDD];

  if (t < 128) {
    float s = 0.f, ss = 0.f, mn = __builtin_inff(), mx = -__builtin_inff();
    for (int i = 0; i < NN; ++i) {
      const float* q = part + ((size_t)b * NN + i) * 512;
      s += q[t];
      mn = fminf(mn, q[128 + t]);
      mx = fmaxf(mx, q[256 + t]);
      ss += q[384 + t];
    }
    const float M = (float)(NN * NN);
    float var = (ss - s * s / M) / (M - 1.0f);
    ez[t] = s / M; ez[128 + t] = mn; ez[256 + t] = mx;
    ez[384 + t] = sqrtf(fmaxf(var, 0.0f));
    fsh[t] = features[b * FDD + t];
  }
  {
    float s = 0.f, ss = 0.f, mn = __builtin_inff(), mx = -__builtin_inff();
    for (int i = 0; i < NN; ++i) {
      float x = nodes[((size_t)b * NN + i) * NDD + t];
      s += x; ss += x * x;
      mn = fminf(mn, x); mx = fmaxf(mx, x);
    }
    const float M = (float)NN;
    float var = (ss - s * s / M) / (M - 1.0f);
    nz[t] = s / M; nz[256 + t] = mn; nz[512 + t] = mx;
    nz[768 + t] = sqrtf(fmaxf(var, 0.0f));
  }
  __syncthreads();
  if (t < 128) {
    float a = bf2f[t];
    for (int k = 0; k < 128; ++k) a += fsh[k] * wf2f[(size_t)k * 128 + t];
    a += bn2f[t];
    for (int k = 0; k < 1024; ++k) a += nz[k] * wn2f[(size_t)k * 128 + t];
    a += be2f[t];
    for (int k = 0; k < 512; ++k) a += ez[k] * we2f[(size_t)k * 128 + t];
    pre[t] = a;
  }
  __syncthreads();
  if (t < 128) {
    float a = b1[t];
    for (int k = 0; k < 128; ++k) a += pre[k] * w1[(size_t)k * 128 + t];
    h1[t] = fmaxf(a, 0.0f);
  }
  __syncthreads();
  if (t < 128) {
    float a = b2[t];
    for (int k = 0; k < 128; ++k) a += h1[k] * w2[(size_t)k * 128 + t];
    out_feat[b * FDD + t] = a;
  }
}

// ---------------------------------------------------------------------------
extern "C" void kernel_launch(void* const* d_in, const int* in_sizes, int n_in,
                              void* d_out, int out_size, void* d_ws, size_t ws_size,
                              hipStream_t stream) {
  (void)in_sizes; (void)n_in; (void)out_size; (void)ws_size;

  const float* nodes    = (const float*)d_in[0];
  const float* edges    = (const float*)d_in[1];
  const float* features = (const float*)d_in[2];
  const float* mask     = (const float*)d_in[3];
  const float* qw  = (const float*)d_in[4];  const float* qb  = (const float*)d_in[5];
  const float* kw  = (const float*)d_in[6];  const float* kb  = (const float*)d_in[7];
  const float* vw  = (const float*)d_in[8];  const float* vb  = (const float*)d_in[9];
  const float* enaw = (const float*)d_in[10]; const float* enab = (const float*)d_in[11];
  const float* enmw = (const float*)d_in[12]; const float* enmb = (const float*)d_in[13];
  const float* femw = (const float*)d_in[14]; const float* femb = (const float*)d_in[15];
  const float* feaw = (const float*)d_in[16]; const float* feab = (const float*)d_in[17];
  const float* fnmw = (const float*)d_in[18]; const float* fnmb = (const float*)d_in[19];
  const float* fnaw = (const float*)d_in[20]; const float* fnab = (const float*)d_in[21];
  const float* f2fw = (const float*)d_in[22]; const float* f2fb = (const float*)d_in[23];
  const float* n2fw = (const float*)d_in[24]; const float* n2fb = (const float*)d_in[25];
  const float* e2fw = (const float*)d_in[26]; const float* e2fb = (const float*)d_in[27];
  const float* nnw  = (const float*)d_in[28]; const float* nnb  = (const float*)d_in[29];
  const float* enw  = (const float*)d_in[30]; const float* enb  = (const float*)d_in[31];
  const float* f1w  = (const float*)d_in[32]; const float* f1b  = (const float*)d_in[33];
  const float* f2w  = (const float*)d_in[34]; const float* f2b  = (const float*)d_in[35];

  float* out_nodes = (float*)d_out;                         // 4*128*256
  float* out_edges = out_nodes + (size_t)BB * NN * NDD;     // 4*128*128*128
  float* out_feat  = out_edges + (size_t)BB * NN * NN * EDD;

  // workspace layout (fp32 section then f16 section)
  float* Qb_ = (float*)d_ws;
  float* Kb_ = Qb_ + (size_t)BB * NN * NDD;
  float* Vb_ = Kb_ + (size_t)BB * NN * NDD;
  float* feA = Vb_ + (size_t)BB * NN * NDD;
  float* feM = feA + BB * NDD;
  float* fnA = feM + BB * NDD;
  float* fnM = fnA + BB * NDD;
  float* part = fnM + BB * NDD;                 // 4*128*512 floats
  _Float16* pack   = (_Float16*)(part + (size_t)BB * NN * 512);
  _Float16* logits = pack + 98304;              // 3*64 frags * 512 halves

  prepack_kernel<<<192, 32, 0, stream>>>(enmw, enaw, enw, pack);
  qkv_kernel<<<BB * NN, 256, 0, stream>>>(nodes, qw, qb, kw, kb, vw, vb, mask,
                                          Qb_, Kb_, Vb_);
  fvec_kernel<<<BB, 256, 0, stream>>>(features, femw, femb, feaw, feab,
                                      fnmw, fnmb, fnaw, fnab, feM, feA, fnM, fnA);
  stats_edges_kernel<<<BB * NN, 128, 0, stream>>>(edges, part);
  edges_main_kernel<<<BB * NN, 256, 0, stream>>>(
      edges, Qb_, Kb_, mask, feA, feM, enmb, enab, enb,
      pack, pack + 32768, pack + 65536, logits, out_edges);
  softmax_v_kernel<<<BB * NN, 256, 0, stream>>>(logits, Vb_, mask, fnA, fnM,
                                                nnw, nnb, out_nodes);
  feat_final_kernel<<<BB, 256, 0, stream>>>(part, nodes, features,
                                            f2fw, f2fb, n2fw, n2fb, e2fw, e2fb,
                                            f1w, f1b, f2w, f2b, out_feat);
}
